// MinimalRNN_24120536334722
// MI455X (gfx1250) — matrix operand order repaired and checked by reading, not compiled, not run on hardware
//
#include <hip/hip_runtime.h>
#include <math.h>

// ---------------- problem constants ----------------
#define SEQ   256
#define BSZ   256
#define HID   1024
#define NLAY  3
#define DEPTH 4                 // ring-buffer pipeline depth (steps)
#define THREADS 512             // 16 waves (wave32)
#define WAVES (THREADS / 32)
#define NTILES (HID / 16)       // 64 N-tiles of 16
#define NT_PER_WAVE (NTILES / WAVES)   // 4
#define KTILES (HID / 32)       // 32 K-tiles of 32 (bf16 WMMA K)
#define SROW  (HID + 8)         // padded LDS row stride (ushorts) -> bank rotation
#define CHUNKS (16 * HID / 8)   // 2048 b128 chunks per 16xHID bf16 tile

typedef __attribute__((ext_vector_type(16))) __bf16 v16bf;
typedef __attribute__((ext_vector_type(8)))  float  v8f;

union Frag { v16bf v; uint4 q[2]; };

__device__ __forceinline__ unsigned short f2bf(float f) {
  unsigned u = __float_as_uint(f);
  u += 0x7FFFu + ((u >> 16) & 1u);        // round-to-nearest-even
  return (unsigned short)(u >> 16);
}
__device__ __forceinline__ unsigned pk2bf(float lo, float hi) {
  return (unsigned)f2bf(lo) | ((unsigned)f2bf(hi) << 16);
}

__device__ __forceinline__ int ld_acq(const int* p) {
  return __hip_atomic_load(p, __ATOMIC_ACQUIRE, __HIP_MEMORY_SCOPE_AGENT);
}
__device__ __forceinline__ void st_rel(int* p, int v) {
  __hip_atomic_store(p, v, __ATOMIC_RELEASE, __HIP_MEMORY_SCOPE_AGENT);
}

// WGP-scope prefetch: scope 0 pulls into ALL cache levels (incl. WGP$/L0).
// SE/DEV/SYS-scope prefetches (what __builtin_prefetch produces) stop at GL2.
__device__ __forceinline__ void prefetch_wgp(const void* p) {
  asm volatile("global_prefetch_b8 %0, off" :: "v"(p) : "memory");
}

// CDNA5 async LDS<->global engine (ASYNCcnt-tracked)
__device__ __forceinline__ void async_ld_lds_b128(unsigned lds_off,
                                                  const void* gaddr) {
  asm volatile("global_load_async_to_lds_b128 %0, %1, off"
               :: "v"(lds_off), "v"(gaddr) : "memory");
}
__device__ __forceinline__ void async_st_lds_b128(void* gaddr,
                                                  unsigned lds_off) {
  asm volatile("global_store_async_from_lds_b128 %0, %1, off scope:SCOPE_DEV"
               :: "v"(gaddr), "v"(lds_off) : "memory");
}
__device__ __forceinline__ void wait_asynccnt0() {
  asm volatile("s_wait_asynccnt 0x0" ::: "memory");
}

// ---------------- weight fp32 -> bf16 prepass ----------------
__global__ void convert_w_kernel(const float* __restrict__ wih,
                                 const float* __restrict__ whh,
                                 unsigned short* __restrict__ o_ih,
                                 unsigned short* __restrict__ o_hh, int n) {
  int i = blockIdx.x * blockDim.x + threadIdx.x;
  if (i < n) {
    o_ih[i] = f2bf(wih[i]);
    o_hh[i] = f2bf(whh[i]);
  }
}

// ---------------- persistent pipelined RNN ----------------
// grid = 48 WGs: WG(bt, l) owns batch rows [bt*16, bt*16+16) of layer l for all t.
__global__ void __launch_bounds__(THREADS)
rnn_pipeline_kernel(const float* __restrict__ x,
                    const float* __restrict__ h0,
                    const float* __restrict__ b_ih,
                    const float* __restrict__ b_hh,
                    const unsigned short* __restrict__ wih_bf,
                    const unsigned short* __restrict__ whh_bf,
                    unsigned short* __restrict__ ring,
                    int* __restrict__ prodf,
                    int* __restrict__ consf,
                    float* __restrict__ y,
                    float* __restrict__ hN) {
  __shared__ __align__(16) unsigned short s_inp[16 * SROW];
  __shared__ __align__(16) unsigned short s_h[2][16 * SROW];

  const int bt  = blockIdx.x & 15;   // batch tile
  const int l   = blockIdx.x >> 4;   // layer 0..2
  const int tid = threadIdx.x;
  const unsigned lane  = tid & 31u;
  const unsigned wv    = tid >> 5;        // wave id 0..15
  const unsigned mrow  = lane & 15u;      // A row / B col / D col
  const unsigned khalf = lane >> 4;       // K half selector

  const unsigned short* wih_l = wih_bf + (size_t)l * HID * HID;
  const unsigned short* whh_l = whh_bf + (size_t)l * HID * HID;

  // stage initial hidden state h0[l, bt*16.., :] into LDS (bf16), vectorized
  for (int i = tid; i < CHUNKS; i += THREADS) {
    const int r = i >> 7, c = (i & 127) * 8;
    const float4* gp =
        (const float4*)(h0 + ((size_t)l * BSZ + bt * 16 + r) * HID + c);
    float4 f0 = gp[0], f1 = gp[1];
    uint4 p = {pk2bf(f0.x, f0.y), pk2bf(f0.z, f0.w),
               pk2bf(f1.x, f1.y), pk2bf(f1.z, f1.w)};
    *(uint4*)&s_h[0][r * SROW + c] = p;
  }
  __syncthreads();

  int cur = 0;
  for (int t = 0; t < SEQ; ++t) {
    // ---- cross-WG flow control (one lane spins, scoped atomics) ----
    if (tid == 0) {
      if (l > 0) { while (ld_acq(&prodf[((l - 1) * 16 + bt) * 8]) < t + 1) {} }
      if (l < 2 && t >= DEPTH) {
        while (ld_acq(&consf[(l * 16 + bt) * 8]) < t + 1 - DEPTH) {}
      }
    }
    __syncthreads();

    // ---- stage this step's layer input into LDS ----
    if (l == 0) {
      // fp32 -> bf16 on the fly, vectorized (float4 x2 -> one b128 LDS store)
      for (int i = tid; i < CHUNKS; i += THREADS) {
        const int r = i >> 7, c = (i & 127) * 8;
        const float4* gp =
            (const float4*)(x + ((size_t)t * BSZ + bt * 16 + r) * HID + c);
        float4 f0 = gp[0], f1 = gp[1];
        uint4 p = {pk2bf(f0.x, f0.y), pk2bf(f0.z, f0.w),
                   pk2bf(f1.x, f1.y), pk2bf(f1.z, f1.w)};
        *(uint4*)&s_inp[r * SROW + c] = p;
      }
    } else {
      // bf16 ring slot -> LDS via the async engine (no VGPR round-trip)
      const unsigned short* slot =
          ring + (((size_t)(l - 1) * 16 + bt) * DEPTH + (t & (DEPTH - 1))) *
                     (size_t)(16 * HID);
      for (int i = tid; i < CHUNKS; i += THREADS) {
        const int r = i >> 7, c = (i & 127) * 8;
        async_ld_lds_b128((unsigned)(uintptr_t)&s_inp[r * SROW + c],
                          slot + (size_t)i * 8);
      }
      wait_asynccnt0();
    }
    __syncthreads();
    if (tid == 0 && l > 0) st_rel(&consf[((l - 1) * 16 + bt) * 8], t + 1);

    // ---- two GEMMs: inp @ W_ih^T  +  h_prev @ W_hh^T ----
    v8f acc[NT_PER_WAVE] = {};
    const unsigned short* sA = s_inp    + mrow * SROW + khalf * 8;
    const unsigned short* sH = s_h[cur] + mrow * SROW + khalf * 8;

#pragma unroll 2
    for (int kt = 0; kt < KTILES; ++kt) {
      const int kb = kt * 32;
      Frag a;
      a.q[0] = *(const uint4*)(sA + kb);
      a.q[1] = *(const uint4*)(sA + kb + 16);
#pragma unroll
      for (int j = 0; j < NT_PER_WAVE; ++j) {
        const unsigned n = (wv * NT_PER_WAVE + j) * 16 + mrow;
        const unsigned short* bp = wih_l + (size_t)n * HID + kb + khalf * 8;
        if (j == 0)  // warm WGP$/L0 for the recurrent GEMM (scope-0 prefetch)
          prefetch_wgp(whh_l + (size_t)n * HID + kb + khalf * 16);
        Frag b;
        b.q[0] = *(const uint4*)bp;
        b.q[1] = *(const uint4*)(bp + 16);
        acc[j] = __builtin_amdgcn_wmma_f32_16x16x32_bf16(
            false, a.v, false, b.v, (short)0, acc[j], false, false);
      }
    }
#pragma unroll 2
    for (int kt = 0; kt < KTILES; ++kt) {
      const int kb = kt * 32;
      Frag a;
      a.q[0] = *(const uint4*)(sH + kb);
      a.q[1] = *(const uint4*)(sH + kb + 16);
#pragma unroll
      for (int j = 0; j < NT_PER_WAVE; ++j) {
        const unsigned n = (wv * NT_PER_WAVE + j) * 16 + mrow;
        const unsigned short* bp = whh_l + (size_t)n * HID + kb + khalf * 8;
        Frag b;
        b.q[0] = *(const uint4*)bp;
        b.q[1] = *(const uint4*)(bp + 16);
        acc[j] = __builtin_amdgcn_wmma_f32_16x16x32_bf16(
            false, a.v, false, b.v, (short)0, acc[j], false, false);
      }
    }

    // ---- epilogue: bias + tanh, write LDS state (+ fp32 outputs) ----
    const int nxt = cur ^ 1;
#pragma unroll
    for (int j = 0; j < NT_PER_WAVE; ++j) {
      const unsigned n = (wv * NT_PER_WAVE + j) * 16 + mrow;
      const float bias = b_ih[l * HID + n] + b_hh[l * HID + n];
#pragma unroll
      for (int r = 0; r < 8; ++r) {
        const unsigned m = khalf * 8 + r;   // D-matrix row per ISA layout
        const float o = tanhf(acc[j][r] + bias);
        s_h[nxt][m * SROW + n] = f2bf(o);
        if (l == 2) y[((size_t)t * BSZ + bt * 16 + m) * HID + n] = o;
        if (t == SEQ - 1)
          hN[((size_t)l * BSZ + bt * 16 + m) * HID + n] = o;
      }
    }
    __syncthreads();   // s_h[nxt] fully written

    if (l < 2) {
      // publish new hidden state to the ring: LDS -> L2 via async engine,
      // SCOPE_DEV so ASYNCcnt==0 implies device visibility before the flag.
      unsigned short* slot_out =
          ring + (((size_t)l * 16 + bt) * DEPTH + (t & (DEPTH - 1))) *
                     (size_t)(16 * HID);
      for (int i = tid; i < CHUNKS; i += THREADS) {
        const int r = i >> 7, c = (i & 127) * 8;
        async_st_lds_b128(slot_out + (size_t)i * 8,
                          (unsigned)(uintptr_t)&s_h[nxt][r * SROW + c]);
      }
      wait_asynccnt0();
      __syncthreads();   // all waves' async stores complete
      if (tid == 0) st_rel(&prodf[(l * 16 + bt) * 8], t + 1);
    }
    cur = nxt;
  }
}

// ---------------- host glue ----------------
extern "C" void kernel_launch(void* const* d_in, const int* in_sizes, int n_in,
                              void* d_out, int out_size, void* d_ws,
                              size_t ws_size, hipStream_t stream) {
  const float* x   = (const float*)d_in[0];
  const float* h0  = (const float*)d_in[1];
  const float* wih = (const float*)d_in[2];
  const float* whh = (const float*)d_in[3];
  const float* bih = (const float*)d_in[4];
  const float* bhh = (const float*)d_in[5];
  float* out = (float*)d_out;

  // workspace layout (bf16 weights | ring | flags)
  unsigned short* ws16   = (unsigned short*)d_ws;
  unsigned short* wih_bf = ws16;
  unsigned short* whh_bf = wih_bf + (size_t)NLAY * HID * HID;
  unsigned short* ring   = whh_bf + (size_t)NLAY * HID * HID;
  const size_t ring_elems = (size_t)2 * 16 * DEPTH * 16 * HID;
  int* flags = (int*)(ring + ring_elems);
  int* prodf = flags;
  int* consf = flags + 2 * 16 * 8;

  hipMemsetAsync(flags, 0, (size_t)2 * (2 * 16 * 8) * sizeof(int), stream);

  const int nw = NLAY * HID * HID;
  convert_w_kernel<<<(nw + 255) / 256, 256, 0, stream>>>(wih, whh, wih_bf,
                                                         whh_bf, nw);

  rnn_pipeline_kernel<<<48, THREADS, 0, stream>>>(
      x, h0, bih, bhh, wih_bf, whh_bf, ring, prodf, consf, out,
      out + (size_t)SEQ * BSZ * HID);
}
